// TimestepEmbedder_42090679501016
// MI455X (gfx1250) — 8-bit matrix operand order repaired and checked by reading, compiled, not run on hardware
//
#include <hip/hip_runtime.h>
#include <hip/hip_bf16.h>
#include <math.h>

typedef int v8i __attribute__((ext_vector_type(8)));
typedef int v4i_g __attribute__((vector_size(16)));

#define HIDDEN 2048
#define FREQ   256
#define INTER  5632
#define BATCH  8192

#define BLK_M 256
#define BLK_N 64
#define LDSROW 80

#if defined(__has_builtin)
#  if __has_builtin(__builtin_amdgcn_global_load_async_to_lds_b128)
#    define HAVE_ASYNC_LDS 1
#  endif
#endif

__device__ __forceinline__ void copy16_to_lds(signed char* lds, const signed char* g)
{
#ifdef HAVE_ASYNC_LDS
  __builtin_amdgcn_global_load_async_to_lds_b128(
      (__attribute__((address_space(1))) v4i_g*)g,
      (__attribute__((address_space(3))) v4i_g*)lds, 0, 0);
#else
  *(int4*)lds = *(const int4*)g;
#endif
}

__device__ __forceinline__ void wait_async_lds()
{
#ifdef HAVE_ASYNC_LDS
#  if __has_builtin(__builtin_amdgcn_s_wait_asynccnt)
  __builtin_amdgcn_s_wait_asynccnt(0);
#  else
  asm volatile("s_wait_asynccnt 0x0" ::: "memory");
#  endif
#endif
}

__global__ __launch_bounds__(256) void embed_rmsq_kernel(
    const float* __restrict__ t, const float* __restrict__ g,
    signed char* __restrict__ xq, float* __restrict__ inv_s)
{
  const int tok = blockIdx.x;
  const int i   = threadIdx.x;
  const float tv = t[tok];

  const int half = FREQ / 2;
  int j = (i < half) ? i : (i - half);
  float freq = __expf(-logf(10000.0f) * (float)j / (float)half);
  float arg  = tv * freq;
  float v    = (i < half) ? __cosf(arg) : __sinf(arg);

  __shared__ float red[256];
  __shared__ float rn_sh, s_sh;

  red[i] = v * v;
  __syncthreads();
  for (int s = 128; s > 0; s >>= 1) { if (i < s) red[i] += red[i + s]; __syncthreads(); }
  if (i == 0) rn_sh = rsqrtf(red[0] / (float)FREQ + 1e-6f);
  __syncthreads();

  float xn = v * rn_sh * g[i];

  red[i] = fabsf(xn);
  __syncthreads();
  for (int s = 128; s > 0; s >>= 1) { if (i < s) red[i] = fmaxf(red[i], red[i + s]); __syncthreads(); }
  if (i == 0) { float m = fmaxf(red[0], 1e-5f); s_sh = 127.0f / m; inv_s[tok] = m / 127.0f; }
  __syncthreads();

  float q = fminf(fmaxf(rintf(xn * s_sh), -128.0f), 127.0f);
  xq[(long)tok * FREQ + i] = (signed char)q;
}

__global__ __launch_bounds__(256) void rmsq_kernel(
    const float* __restrict__ x, const float* __restrict__ g,
    signed char* __restrict__ xq, float* __restrict__ inv_s, int L)
{
  const int row = blockIdx.x;
  const float* xr = x + (long)row * L;
  __shared__ float red[256];
  __shared__ float rn_sh, s_sh;

  float ss = 0.f, mx = 0.f;
  for (int i = threadIdx.x; i < L; i += 256) {
    float v = xr[i];
    ss += v * v;
    mx = fmaxf(mx, fabsf(v * g[i]));
  }
  red[threadIdx.x] = ss; __syncthreads();
  for (int s = 128; s > 0; s >>= 1) { if (threadIdx.x < s) red[threadIdx.x] += red[threadIdx.x + s]; __syncthreads(); }
  if (threadIdx.x == 0) rn_sh = rsqrtf(red[0] / (float)L + 1e-6f);
  __syncthreads();

  red[threadIdx.x] = mx; __syncthreads();
  for (int s = 128; s > 0; s >>= 1) { if (threadIdx.x < s) red[threadIdx.x] = fmaxf(red[threadIdx.x], red[threadIdx.x + s]); __syncthreads(); }
  if (threadIdx.x == 0) { float m = fmaxf(red[0] * rn_sh, 1e-5f); s_sh = 127.0f / m; inv_s[row] = m / 127.0f; }
  __syncthreads();

  float sc = rn_sh * s_sh;
  for (int i = threadIdx.x; i < L; i += 256) {
    float q = fminf(fmaxf(rintf(xr[i] * g[i] * sc), -128.0f), 127.0f);
    xq[(long)row * L + i] = (signed char)q;
  }
}

__global__ __launch_bounds__(256) void absmean_partial_kernel(
    const float* __restrict__ w, long n, float* __restrict__ partial)
{
  __shared__ float red[256];
  float acc = 0.f;
  for (long i = (long)blockIdx.x * 256 + threadIdx.x; i < n; i += (long)gridDim.x * 256)
    acc += fabsf(w[i]);
  red[threadIdx.x] = acc; __syncthreads();
  for (int s = 128; s > 0; s >>= 1) { if (threadIdx.x < s) red[threadIdx.x] += red[threadIdx.x + s]; __syncthreads(); }
  if (threadIdx.x == 0) partial[blockIdx.x] = red[0];
}

__global__ __launch_bounds__(256) void finalize_scales_kernel(
    const float* __restrict__ partials, float* __restrict__ scales,
    float n0, float n1, float n2)
{
  __shared__ float red[256];
  float cnt[3] = { n0, n1, n2 };
  for (int tsr = 0; tsr < 3; ++tsr) {
    red[threadIdx.x] = partials[tsr * 256 + threadIdx.x];
    __syncthreads();
    for (int s = 128; s > 0; s >>= 1) { if (threadIdx.x < s) red[threadIdx.x] += red[threadIdx.x + s]; __syncthreads(); }
    if (threadIdx.x == 0) {
      float mean = red[0] / cnt[tsr];
      float dq = fmaxf(mean, 1e-5f);
      scales[tsr]     = dq;
      scales[3 + tsr] = 1.0f / dq;
    }
    __syncthreads();
  }
}

__global__ __launch_bounds__(256) void quant_weight_kernel(
    const float* __restrict__ w, signed char* __restrict__ wq, long n,
    const float* __restrict__ scales, int idx)
{
  const float s = scales[3 + idx];
  for (long i = (long)blockIdx.x * 256 + threadIdx.x; i < n; i += (long)gridDim.x * 256) {
    float q = fminf(fmaxf(rintf(w[i] * s), -1.0f), 1.0f);
    wq[i] = (signed char)q;
  }
}

__device__ __forceinline__ v8i load_a_frag(const signed char* ap)
{
  v8i a;
  int2 c0 = *(const int2*)(ap + 0);
  int2 c1 = *(const int2*)(ap + 16);
  int2 c2 = *(const int2*)(ap + 32);
  int2 c3 = *(const int2*)(ap + 48);
  a[0] = c0.x; a[1] = c0.y; a[2] = c1.x; a[3] = c1.y;
  a[4] = c2.x; a[5] = c2.y; a[6] = c3.x; a[7] = c3.y;
  return a;
}

__device__ __forceinline__ v8i load_b_frag(const signed char* bp)
{
  v8i b;
  int2 b0 = *(const int2*)(bp + 0);
  int2 b1 = *(const int2*)(bp + 16);
  int2 b2 = *(const int2*)(bp + 32);
  int2 b3 = *(const int2*)(bp + 48);
  b[0] = b0.x; b[1] = b0.y; b[2] = b1.x; b[3] = b1.y;
  b[4] = b2.x; b[5] = b2.y; b[6] = b3.x; b[7] = b3.y;
  return b;
}

__device__ __forceinline__ void gemm_step(const signed char* ldsbase, int l16, int h,
                                          const v8i& af0, const v8i& af1, v8i acc[2][4])
{
  v8i bf[4];
  #pragma unroll
  for (int nt = 0; nt < 4; ++nt)
    bf[nt] = load_b_frag(ldsbase + (nt * 16 + l16) * LDSROW + 8 * h);
  #pragma unroll
  for (int nt = 0; nt < 4; ++nt) {
    acc[0][nt] = __builtin_amdgcn_wmma_i32_16x16x64_iu8(
        true, af0, true, bf[nt], acc[0][nt], false, false);
    acc[1][nt] = __builtin_amdgcn_wmma_i32_16x16x64_iu8(
        true, af1, true, bf[nt], acc[1][nt], false, false);
  }
}

__global__ __launch_bounds__(256) void bit_gemm_kernel(
    const signed char* __restrict__ A,
    const float* __restrict__ inv_sa,
    const signed char* __restrict__ Wq,
    const float* __restrict__ scales, int widx,
    float* __restrict__ out,
    int N, int K)
{
  __shared__ signed char ldsB[2][BLK_N * LDSROW];
  const int tid  = threadIdx.x;
  const int wave = tid >> 5;
  const int lane = tid & 31;
  const int h    = lane >> 4;
  const int l16  = lane & 15;
  const int m_base = blockIdx.y * BLK_M + wave * 32;
  const int n_base = blockIdx.x * BLK_N;
  const float wdq  = scales[widx];

  v8i acc[2][4];
  #pragma unroll
  for (int mi = 0; mi < 2; ++mi)
    #pragma unroll
    for (int nt = 0; nt < 4; ++nt) acc[mi][nt] = (v8i){0,0,0,0,0,0,0,0};

  const signed char* a0 = A + (long)(m_base + l16) * K + 8 * h;
  const signed char* a1 = A + (long)(m_base + 16 + l16) * K + 8 * h;
  const int sr = tid >> 2;
  const int sc = (tid & 3) * 16;
  const signed char* wrow = Wq + (long)(n_base + sr) * K + sc;
  signed char* lds0 = &ldsB[0][sr * LDSROW + sc];
  signed char* lds1 = &ldsB[1][sr * LDSROW + sc];

  copy16_to_lds(lds0, wrow);
  v8i afx0 = load_a_frag(a0);
  v8i afx1 = load_a_frag(a1);
  v8i afy0, afy1;

  for (int k0 = 0; k0 < K; k0 += 128) {
    wait_async_lds();
    __syncthreads();
    copy16_to_lds(lds1, wrow + k0 + 64);
    afy0 = load_a_frag(a0 + k0 + 64);
    afy1 = load_a_frag(a1 + k0 + 64);
    if (k0 + 128 < K) __builtin_prefetch((const void*)(wrow + k0 + 128), 0, 1);
    gemm_step(ldsB[0], l16, h, afx0, afx1, acc);

    wait_async_lds();
    __syncthreads();
    if (k0 + 128 < K) {
      copy16_to_lds(lds0, wrow + k0 + 128);
      afx0 = load_a_frag(a0 + k0 + 128);
      afx1 = load_a_frag(a1 + k0 + 128);
    }
    gemm_step(ldsB[1], l16, h, afy0, afy1, acc);
  }

  #pragma unroll
  for (int mi = 0; mi < 2; ++mi)
    #pragma unroll
    for (int nt = 0; nt < 4; ++nt)
      #pragma unroll
      for (int r = 0; r < 8; ++r) {
        int row = m_base + mi * 16 + 8 * h + r;
        int col = n_base + nt * 16 + l16;
        out[(long)row * N + col] = (float)acc[mi][nt][r] * inv_sa[row] * wdq;
      }
}

__global__ __launch_bounds__(256) void bit_gemm_swiglu_kernel(
    const signed char* __restrict__ A, const float* __restrict__ inv_sa,
    const signed char* __restrict__ Wq, const float* __restrict__ scales, int widx,
    float* __restrict__ outh,
    int Nh, int K)
{
  __shared__ signed char ldsG[2][BLK_N * LDSROW];
  __shared__ signed char ldsV[2][BLK_N * LDSROW];
  const int tid  = threadIdx.x;
  const int wave = tid >> 5;
  const int lane = tid & 31;
  const int h    = lane >> 4;
  const int l16  = lane & 15;
  const int m_base = blockIdx.y * BLK_M + wave * 32;
  const int n_base = blockIdx.x * BLK_N;
  const float wdq  = scales[widx];

  v8i accg[2][4], accv[2][4];
  #pragma unroll
  for (int mi = 0; mi < 2; ++mi)
    #pragma unroll
    for (int nt = 0; nt < 4; ++nt) {
      accg[mi][nt] = (v8i){0,0,0,0,0,0,0,0};
      accv[mi][nt] = (v8i){0,0,0,0,0,0,0,0};
    }

  const signed char* a0 = A + (long)(m_base + l16) * K + 8 * h;
  const signed char* a1 = A + (long)(m_base + 16 + l16) * K + 8 * h;
  const int sr = tid >> 2;
  const int sc = (tid & 3) * 16;
  const signed char* wrowG = Wq + (long)(n_base + sr) * K + sc;
  const signed char* wrowV = Wq + (long)(Nh + n_base + sr) * K + sc;
  signed char* ldsg0 = &ldsG[0][sr * LDSROW + sc];
  signed char* ldsg1 = &ldsG[1][sr * LDSROW + sc];
  signed char* ldsv0 = &ldsV[0][sr * LDSROW + sc];
  signed char* ldsv1 = &ldsV[1][sr * LDSROW + sc];

  copy16_to_lds(ldsg0, wrowG);
  copy16_to_lds(ldsv0, wrowV);
  v8i afx0 = load_a_frag(a0);
  v8i afx1 = load_a_frag(a1);
  v8i afy0, afy1;

  for (int k0 = 0; k0 < K; k0 += 128) {
    wait_async_lds();
    __syncthreads();
    copy16_to_lds(ldsg1, wrowG + k0 + 64);
    copy16_to_lds(ldsv1, wrowV + k0 + 64);
    afy0 = load_a_frag(a0 + k0 + 64);
    afy1 = load_a_frag(a1 + k0 + 64);
    if (k0 + 128 < K) {
      __builtin_prefetch((const void*)(wrowG + k0 + 128), 0, 1);
      __builtin_prefetch((const void*)(wrowV + k0 + 128), 0, 1);
    }
    gemm_step(ldsG[0], l16, h, afx0, afx1, accg);
    gemm_step(ldsV[0], l16, h, afx0, afx1, accv);

    wait_async_lds();
    __syncthreads();
    if (k0 + 128 < K) {
      copy16_to_lds(ldsg0, wrowG + k0 + 128);
      copy16_to_lds(ldsv0, wrowV + k0 + 128);
      afx0 = load_a_frag(a0 + k0 + 128);
      afx1 = load_a_frag(a1 + k0 + 128);
    }
    gemm_step(ldsG[1], l16, h, afy0, afy1, accg);
    gemm_step(ldsV[1], l16, h, afy0, afy1, accv);
  }

  #pragma unroll
  for (int mi = 0; mi < 2; ++mi)
    #pragma unroll
    for (int nt = 0; nt < 4; ++nt)
      #pragma unroll
      for (int r = 0; r < 8; ++r) {
        int row = m_base + mi * 16 + 8 * h + r;
        int col = n_base + nt * 16 + l16;
        float dq = inv_sa[row] * wdq;
        float gv = (float)accg[mi][nt][r] * dq;
        float vv = (float)accv[mi][nt][r] * dq;
        float sig = 1.0f / (1.0f + __expf(-gv));
        outh[(long)row * Nh + col] = gv * sig * vv;
      }
}

extern "C" void kernel_launch(void* const* d_in, const int* in_sizes, int n_in,
                              void* d_out, int out_size, void* d_ws, size_t ws_size,
                              hipStream_t stream)
{
  (void)in_sizes; (void)n_in; (void)out_size; (void)ws_size;

  const float* t      = (const float*)d_in[0];
  const float* w_in   = (const float*)d_in[1];
  const float* g_in   = (const float*)d_in[2];
  const float* w_gate = (const float*)d_in[3];
  const float* g_gate = (const float*)d_in[4];
  const float* w_down = (const float*)d_in[5];
  const float* g_down = (const float*)d_in[6];
  float* out = (float*)d_out;

  char* w = (char*)d_ws;
  size_t off = 0;
  auto take = [&](size_t b) { size_t r = off; off += (b + 255) & ~(size_t)255; return r; };

  signed char* xq0     = (signed char*)(w + take((size_t)BATCH * FREQ));
  float*       inv_s0  = (float*)      (w + take((size_t)BATCH * 4));
  signed char* wq_in   = (signed char*)(w + take((size_t)HIDDEN * FREQ));
  signed char* wq_gate = (signed char*)(w + take((size_t)2 * INTER * HIDDEN));
  signed char* wq_down = (signed char*)(w + take((size_t)HIDDEN * INTER));
  float*       x1      = (float*)      (w + take((size_t)BATCH * HIDDEN * 4));
  signed char* xq1     = (signed char*)(w + take((size_t)BATCH * HIDDEN));
  float*       inv_s1  = (float*)      (w + take((size_t)BATCH * 4));
  float*       hbuf    = (float*)      (w + take((size_t)BATCH * INTER * 4));
  signed char* hq      = (signed char*)(w + take((size_t)BATCH * INTER));
  float*       inv_s2  = (float*)      (w + take((size_t)BATCH * 4));
  float*       partials= (float*)      (w + take((size_t)3 * 256 * 4));
  float*       scales  = (float*)      (w + take((size_t)8 * 4));

  const long n_in_w   = (long)HIDDEN * FREQ;
  const long n_gate_w = (long)2 * INTER * HIDDEN;
  const long n_down_w = (long)HIDDEN * INTER;

  absmean_partial_kernel<<<256, 256, 0, stream>>>(w_in,   n_in_w,   partials + 0 * 256);
  absmean_partial_kernel<<<256, 256, 0, stream>>>(w_gate, n_gate_w, partials + 1 * 256);
  absmean_partial_kernel<<<256, 256, 0, stream>>>(w_down, n_down_w, partials + 2 * 256);
  finalize_scales_kernel<<<1, 256, 0, stream>>>(partials, scales,
      (float)n_in_w, (float)n_gate_w, (float)n_down_w);
  quant_weight_kernel<<<512,  256, 0, stream>>>(w_in,   wq_in,   n_in_w,   scales, 0);
  quant_weight_kernel<<<2048, 256, 0, stream>>>(w_gate, wq_gate, n_gate_w, scales, 1);
  quant_weight_kernel<<<2048, 256, 0, stream>>>(w_down, wq_down, n_down_w, scales, 2);

  embed_rmsq_kernel<<<BATCH, 256, 0, stream>>>(t, g_in, xq0, inv_s0);

  {
    dim3 grid(HIDDEN / BLK_N, BATCH / BLK_M);
    bit_gemm_kernel<<<grid, 256, 0, stream>>>(xq0, inv_s0, wq_in, scales, 0,
                                              x1, HIDDEN, FREQ);
  }

  rmsq_kernel<<<BATCH, 256, 0, stream>>>(x1, g_gate, xq1, inv_s1, HIDDEN);

  {
    dim3 grid(INTER / BLK_N, BATCH / BLK_M);
    bit_gemm_swiglu_kernel<<<grid, 256, 0, stream>>>(xq1, inv_s1, wq_gate, scales, 1,
                                                     hbuf, INTER, HIDDEN);
  }

  rmsq_kernel<<<BATCH, 256, 0, stream>>>(hbuf, g_down, hq, inv_s2, INTER);

  {
    dim3 grid(HIDDEN / BLK_N, BATCH / BLK_M);
    bit_gemm_kernel<<<grid, 256, 0, stream>>>(hq, inv_s2, wq_down, scales, 2,
                                              out, HIDDEN, INTER);
  }
}
